// RelativePositionSDPA_33732673143813
// MI455X (gfx1250) — matrix operand order repaired and checked by reading, not compiled, not run on hardware
//
#include <hip/hip_runtime.h>

typedef __attribute__((ext_vector_type(16))) __bf16 v16bf;
typedef __attribute__((ext_vector_type(8)))  __bf16 v8bf;
typedef __attribute__((ext_vector_type(8)))  float  v8f;
typedef __attribute__((ext_vector_type(4)))  float  v4f;

static __device__ __forceinline__ v8f wmma_bf16(v16bf a, v16bf b, v8f c) {
  return __builtin_amdgcn_wmma_f32_16x16x32_bf16(false, a, false, b, (short)0, c, false, false);
}

static __device__ __forceinline__ void wait_ds() {
  asm volatile("s_wait_dscnt 0x0" ::: "memory");
}

// Load one 16x32 bf16 A-tile (M=16 rows, K=32) from an f32 row-major matrix with
// row stride 64 floats. 'row' is this lane's source row (M = lane&15), c0 the
// starting K column, hh = lane>>4.
// A layout (16-bit, 16x32): lane<16 holds K = {c0+0..7, c0+16..23} of row M,
// lane>=16 holds K = {c0+8..15, c0+24..31}.
static __device__ __forceinline__ v16bf loadA64(const float* __restrict__ base,
                                                int row, int c0, int hh) {
  const float* p = base + (size_t)row * 64 + c0 + 8 * hh;
  v4f x0 = *(const v4f*)(p + 0);
  v4f x1 = *(const v4f*)(p + 4);
  v4f x2 = *(const v4f*)(p + 16);
  v4f x3 = *(const v4f*)(p + 20);
  v16bf a;
#pragma unroll
  for (int t = 0; t < 4; ++t) {
    a[t]      = (__bf16)x0[t];
    a[4 + t]  = (__bf16)x1[t];
    a[8 + t]  = (__bf16)x2[t];
    a[12 + t] = (__bf16)x3[t];
  }
  return a;
}

// One workgroup = 4 independent waves. Wave w handles query rows
// [i0, i0+16) of head-instance n, streaming KV in 32-wide tiles with online
// softmax. Scores are computed transposed (M = kv, N = qrow) so K/R tiles are
// contiguous A-loads and Qu^T/Qv^T live in registers as B for the whole kernel.
__global__ __launch_bounds__(128, 1) void txl_rel_attn(
    const float* __restrict__ Q, const float* __restrict__ K,
    const float* __restrict__ V, const float* __restrict__ UB,
    const float* __restrict__ VB, const float* __restrict__ R,
    float* __restrict__ OUT) {
  constexpr int S = 1024;
  constexpr int D = 64;

  const int tid  = threadIdx.x;
  const int lane = tid & 31;
  const int w    = tid >> 5;
  const int a_l  = lane & 15;   // N index (query row within strip) in C layout
  const int hh   = lane >> 4;   // lane half
  const int n    = blockIdx.x >> 4;  // 0..63 (batch*heads)
  const int qb   = blockIdx.x & 15;  // query block 0..15
  const int i0   = qb * 64 + w * 16; // strip base row
  const int h    = n & 15;           // head index for R / biases

  // Per-wave LDS: GL band 48x16 f32, GG band 48x16 f32, 16 stats floats, P^T bf16.
  __shared__ float ldsF[4][48 * 16 * 2 + 16];
  __shared__ __align__(16) __bf16 ldsPall[4][16 * 32];
  float*  ldsGL = ldsF[w];
  float*  ldsGG = ldsGL + 48 * 16;
  float*  ldsSt = ldsGG + 48 * 16;
  __bf16* ldsP  = ldsPall[w];

  const float* qn = Q + (size_t)n * S * D;
  const float* kn = K + (size_t)n * S * D;
  const float* vn = V + (size_t)n * S * D;
  const float* Rh = R + (size_t)h * 2048 * D;

  // ---- Build persistent B-matrix tiles: Qu^T, Qv^T, Qv'(+1 row)^T ----
  // B layout (bf16, 32x16): lane holds column lane&15; element s = row k (table order: 8*hh+s, 16+8*hh+s-8).
  v16bf quT[2], qvT[2], qvpT[2];
  {
    const float scale = 0.125f;  // 64^-0.5
#pragma unroll
    for (int kb = 0; kb < 2; ++kb) {
      v16bf tu, tv, tp;
#pragma unroll
      for (int nc = 0; nc < 16; ++nc) {
        const int   dIdx = kb * 32 + (nc < 8 ? 8 * hh + nc : 16 + 8 * hh + (nc - 8));
        const float ubv  = UB[h * 64 + dIdx];
        const float vbv  = VB[h * 64 + dIdx];
        float q0 = qn[(size_t)(i0 + a_l) * D + dIdx] * scale;
        tu[nc] = (__bf16)(q0 + ubv);
        tv[nc] = (__bf16)(q0 + vbv);
        int rp = i0 + a_l + 1;              // shifted strip for garbage region
        if (rp > S - 1) rp = S - 1;        // clamped row never selected
        float q1 = qn[(size_t)rp * D + dIdx] * scale;
        tp[nc] = (__bf16)(q1 + vbv);
      }
      quT[kb] = tu; qvT[kb] = tv; qvpT[kb] = tp;
    }
  }

  v8f acc[4];
#pragma unroll
  for (int nt = 0; nt < 4; ++nt) acc[nt] = (v8f){};
  float mRun = -__builtin_inff();
  float lRun = 0.f;

  for (int j0 = 0; j0 < S; j0 += 32) {
    const int relBase = i0 - j0;
    float sc[16];

    // ---- content scores S^T = K_tile @ Qu^T ----
#pragma unroll
    for (int mt = 0; mt < 2; ++mt) {
      v16bf aK0 = loadA64(kn, j0 + 16 * mt + a_l, 0, hh);
      v16bf aK1 = loadA64(kn, j0 + 16 * mt + a_l, 32, hh);
      v8f c = (v8f){};
      c = wmma_bf16(aK0, quT[0], c);
      c = wmma_bf16(aK1, quT[1], c);
#pragma unroll
      for (int vv = 0; vv < 8; ++vv) sc[mt * 8 + vv] = c[vv];
    }

    // ---- positional bands G^T[p][a] = R[p] . qv[a] ----
    if (j0 <= i0 + 15) {  // legit band: p = N-1+j-i, needed only if any j<=i
      const int pL0 = (S - 1) + j0 - i0 - 15;
#pragma unroll
      for (int t = 0; t < 3; ++t) {
        int row = pL0 + 16 * t + a_l;
        row = row < 0 ? 0 : (row > 2047 ? 2047 : row);
        v16bf a0 = loadA64(Rh, row, 0, hh);
        v16bf a1 = loadA64(Rh, row, 32, hh);
        v8f c = (v8f){};
        c = wmma_bf16(a0, qvT[0], c);
        c = wmma_bf16(a1, qvT[1], c);
#pragma unroll
        for (int vv = 0; vv < 8; ++vv)
          ldsGL[(16 * t + vv + 8 * hh) * 16 + a_l] = c[vv];
      }
    }
    if (j0 + 31 >= i0 + 2) {  // garbage band: p = j-i-2, rows i+1
      const int pG0 = j0 - i0 - 17;
#pragma unroll
      for (int t = 0; t < 3; ++t) {
        int row = pG0 + 16 * t + a_l;
        row = row < 0 ? 0 : (row > 2047 ? 2047 : row);
        v16bf a0 = loadA64(Rh, row, 0, hh);
        v16bf a1 = loadA64(Rh, row, 32, hh);
        v8f c = (v8f){};
        c = wmma_bf16(a0, qvpT[0], c);
        c = wmma_bf16(a1, qvpT[1], c);
#pragma unroll
        for (int vv = 0; vv < 8; ++vv)
          ldsGG[(16 * t + vv + 8 * hh) * 16 + a_l] = c[vv];
      }
    }
    wait_ds();

    // ---- add shifted pos scores + online softmax (column a = lane&15) ----
    float mloc = -__builtin_inff();
#pragma unroll
    for (int mt = 0; mt < 2; ++mt)
#pragma unroll
      for (int vv = 0; vv < 8; ++vv) {
        const int m   = 16 * mt + vv + 8 * hh;  // kv offset within tile
        const int rel = m - a_l;                // j - i + (j0 - i0) offsetted
        const int idx = (rel + 15) * 16 + a_l;  // same index for both bands
        float pos = (rel <= relBase) ? ldsGL[idx] : ldsGG[idx];
        if (rel == relBase + 1) pos = 0.f;      // j == i+1 -> zero pad slot
        float s = sc[mt * 8 + vv] + pos;
        sc[mt * 8 + vv] = s;
        mloc = fmaxf(mloc, s);
      }
    mloc = fmaxf(mloc, __shfl_xor(mloc, 16, 32));
    const float mNew  = fmaxf(mRun, mloc);
    const float alpha = __expf(mRun - mNew);  // first tile: exp(-inf) = 0
    float ls = 0.f;
#pragma unroll
    for (int t = 0; t < 16; ++t) {
      float p = __expf(sc[t] - mNew);
      sc[t] = p;
      ls += p;
    }
    ls += __shfl_xor(ls, 16, 32);
    lRun = lRun * alpha + ls;
    mRun = mNew;

    // ---- spill P^T (bf16) + alpha stats ----
#pragma unroll
    for (int mt = 0; mt < 2; ++mt) {
      v8bf pk;
#pragma unroll
      for (int vv = 0; vv < 8; ++vv) pk[vv] = (__bf16)sc[mt * 8 + vv];
      *(v8bf*)(ldsP + a_l * 32 + 16 * mt + 8 * hh) = pk;
    }
    if (lane < 16) ldsSt[lane] = alpha;
    wait_ds();

    // ---- reload P in A-layout (lane = qrow a, regs = kv) ----
    v16bf aP;
    {
      v8bf lo = *(const v8bf*)(ldsP + a_l * 32 + 8 * hh);
      v8bf hi = *(const v8bf*)(ldsP + a_l * 32 + 16 + 8 * hh);
#pragma unroll
      for (int t = 0; t < 8; ++t) { aP[t] = lo[t]; aP[8 + t] = hi[t]; }
    }
    float fac[8];
#pragma unroll
    for (int vv = 0; vv < 8; ++vv) fac[vv] = ldsSt[vv + 8 * hh];

    // ---- O = O*alpha + P @ V_tile (B = V rows, lane = d column, regs = kv) ----
#pragma unroll
    for (int nt = 0; nt < 4; ++nt) {
      v16bf bV;
#pragma unroll
      for (int t = 0; t < 8; ++t) {
        bV[t]     = (__bf16)vn[(size_t)(j0 + 8 * hh + t) * D + 16 * nt + a_l];
        bV[8 + t] = (__bf16)vn[(size_t)(j0 + 16 + 8 * hh + t) * D + 16 * nt + a_l];
      }
      v8f c = acc[nt];
#pragma unroll
      for (int vv = 0; vv < 8; ++vv) c[vv] *= fac[vv];
      acc[nt] = wmma_bf16(aP, bV, c);
    }
  }

  // ---- normalize by l and store (O in normal C layout: M=a, N=d) ----
  if (lane < 16) ldsSt[lane] = 1.f / lRun;
  wait_ds();
  float rf[8];
#pragma unroll
  for (int vv = 0; vv < 8; ++vv) rf[vv] = ldsSt[vv + 8 * hh];
  float* op = OUT + (size_t)n * S * D;
#pragma unroll
  for (int nt = 0; nt < 4; ++nt)
#pragma unroll
    for (int vv = 0; vv < 8; ++vv)
      op[(size_t)(i0 + vv + 8 * hh) * D + 16 * nt + a_l] = acc[nt][vv] * rf[vv];
}

extern "C" void kernel_launch(void* const* d_in, const int* in_sizes, int n_in,
                              void* d_out, int out_size, void* d_ws, size_t ws_size,
                              hipStream_t stream) {
  (void)in_sizes; (void)n_in; (void)out_size; (void)d_ws; (void)ws_size;
  const float* q  = (const float*)d_in[0];
  const float* k  = (const float*)d_in[1];
  const float* v  = (const float*)d_in[2];
  const float* ub = (const float*)d_in[3];
  const float* vb = (const float*)d_in[4];
  const float* R  = (const float*)d_in[5];
  float* out = (float*)d_out;
  // 64 head-instances x 16 query blocks of 64 rows; 4 waves of 16 rows each.
  dim3 grid(64 * 16), block(128);
  hipLaunchKernelGGL(txl_rel_attn, grid, block, 0, stream, q, k, v, ub, vb, R, out);
}
